// AttentionHead_72971494359064
// MI455X (gfx1250) — matrix operand order repaired and checked by reading, not compiled, not run on hardware
//
#include <hip/hip_runtime.h>
#include <math.h>

#define B_  8
#define T_  4096
#define DM  512
#define DK  64

typedef _Float16 v16h __attribute__((ext_vector_type(16)));
typedef _Float16 h8   __attribute__((ext_vector_type(8)));
typedef float    v8f  __attribute__((ext_vector_type(8)));
typedef float    f4   __attribute__((ext_vector_type(4)));
typedef unsigned int u32x4 __attribute__((ext_vector_type(4)));
typedef int          i32x4 __attribute__((ext_vector_type(4)));
typedef int          i32x8 __attribute__((ext_vector_type(8)));

static __device__ __forceinline__ v16h make_frag(h8 lo, h8 hi) {
    v16h a;
#pragma unroll
    for (int i = 0; i < 8; ++i) { a[i] = lo[i]; a[8 + i] = hi[i]; }
    return a;
}

// One wave-wide async DMA of 32 lanes x 16B : global -> LDS (ASYNCcnt-tracked)
static __device__ __forceinline__ void async_b128(_Float16* lds_dst, const _Float16* g_src) {
    uint32_t loff = (uint32_t)(uintptr_t)lds_dst;          // low 32b of generic = LDS offset
    uint64_t gad  = (uint64_t)(uintptr_t)g_src;
    asm volatile("global_load_async_to_lds_b128 %0, %1, off"
                 :: "v"(loff), "v"(gad) : "memory");
}

// TDM: one 2-D tile (16-bit elements) global -> LDS, TENSORcnt-tracked.
// tile_d1 == 0 -> 1-D tile.  Issued once per calling wave.
static __device__ __forceinline__ void tdm_load_2d(
    uint32_t lds_byte, uint64_t gaddr,
    uint32_t tensor_d0, uint32_t tensor_d1,
    uint32_t tile_d0,  uint32_t tile_d1, uint32_t d0_stride)
{
    u32x4 g0;
    g0[0] = 1u;                                        // count=1, user descriptor
    g0[1] = lds_byte;                                  // lds_addr (bytes)
    g0[2] = (uint32_t)gaddr;                           // global_addr[31:0]
    g0[3] = (uint32_t)(gaddr >> 32) | (2u << 30);      // global_addr[56:32] | type=2
    i32x8 g1;
    g1[0] = (int)(1u << 16);                           // data_size=2B, mask=0
    g1[1] = (int)((tensor_d0 & 0xFFFFu) << 16);        // tensor_dim0[15:0] @ b48
    g1[2] = (int)(((tensor_d0 >> 16) & 0xFFFFu) |      // tensor_dim0[31:16]
                  ((tensor_d1 & 0xFFFFu) << 16));      // tensor_dim1[15:0]
    g1[3] = (int)(((tensor_d1 >> 16) & 0xFFFFu) |
                  (tile_d0 << 16));                    // tile_dim0 @ b112
    g1[4] = (int)tile_d1;                              // tile_dim1, tile_dim2=0
    g1[5] = (int)d0_stride;                            // tensor_dim0_stride[31:0]
    g1[6] = 0;                                         // stride[47:32], dim1_stride lo
    g1[7] = 0;
    i32x4 z4 = {0, 0, 0, 0};                           // groups 2/3: <=2-D tensor
    i32x8 z8 = {0, 0, 0, 0, 0, 0, 0, 0};               // unused trailing group
    __builtin_amdgcn_tensor_load_to_lds(g0, g1, z4, z4, z8, 0);
}

// ---------------------------------------------------------------------------
// Kernel 0: convert W [512][64] f32 -> Wt [64][512] f16 (transposed, per z)
// ---------------------------------------------------------------------------
__global__ __launch_bounds__(256) void wcvt_kernel(
    const float* __restrict__ Wk, const float* __restrict__ Wq,
    const float* __restrict__ Wv, _Float16* __restrict__ Wt)
{
    const float* W = (blockIdx.x == 0) ? Wk : (blockIdx.x == 1) ? Wq : Wv;
    _Float16* O = Wt + (size_t)blockIdx.x * DK * DM;
    for (int i = threadIdx.x; i < DM * DK; i += 256) {
        int kk = i >> 6, n = i & 63;
        O[n * DM + kk] = (_Float16)W[i];
    }
}

// ---------------------------------------------------------------------------
// Kernel 1: X[BT,512] @ W[512,64] -> f16, one 16x16 tile per wave.
// z=0: k->Kh (row major), z=1: q->Qh (row major), z=2: v->Vt (TRANSPOSED)
// ---------------------------------------------------------------------------
__global__ __launch_bounds__(128) void proj_kernel(
    const float* __restrict__ kin, const float* __restrict__ qin, const float* __restrict__ vin,
    const _Float16* __restrict__ Wt,
    _Float16* __restrict__ Kh, _Float16* __restrict__ Qh, _Float16* __restrict__ Vt)
{
    const int z = blockIdx.z;
    const float* X = (z == 0) ? kin : (z == 1) ? qin : vin;

    const int lane  = threadIdx.x & 31;
    const int wave  = threadIdx.x >> 5;
    const int tile  = blockIdx.x * 4 + wave;   // (BT/16)*(64/16) = 8192 tiles
    const int tileM = tile >> 2;
    const int tileN = tile & 3;
    const int h     = lane >> 4;
    const int mn    = lane & 15;
    const int row   = tileM * 16 + mn;         // A row
    const int ncol  = tileN * 16 + mn;         // B/D column (0..63)

    const float*    xrow = X + (size_t)row * DM;
    const _Float16* wrow = Wt + ((size_t)z * DK + ncol) * DM;   // Wt[z][ncol][*]
    v8f acc = {};

    for (int kk = 0; kk < DM; kk += 32) {
        // A fragment: two contiguous 8-float runs per lane, convert to f16
        const f4* p0 = (const f4*)(xrow + kk + h * 8);
        const f4* p1 = (const f4*)(xrow + kk + 16 + h * 8);
        f4 a0 = p0[0], a1 = p0[1], a2 = p1[0], a3 = p1[1];
        v16h a;
#pragma unroll
        for (int i = 0; i < 4; ++i) {
            a[i]      = (_Float16)a0[i];
            a[4 + i]  = (_Float16)a1[i];
            a[8 + i]  = (_Float16)a2[i];
            a[12 + i] = (_Float16)a3[i];
        }
        // B fragment: Wt row is K-contiguous -> two b128 loads
        const h8* wb = (const h8*)(wrow + kk + h * 8);
        v16h bm = make_frag(wb[0], wb[2]);

        acc = __builtin_amdgcn_wmma_f32_16x16x32_f16(false, a, false, bm,
                                                     (short)0, acc, false, false);
    }

    if (z == 2) {
        // D rows r+8h are CONTIGUOUS t for fixed feature col -> one 16B store
        int m0 = tileM * 16 + 8 * h;           // global row of acc[0]
        int bb = m0 >> 12;                     // /T_
        int t0 = m0 & (T_ - 1);
        h8 vals;
#pragma unroll
        for (int r = 0; r < 8; ++r) vals[r] = (_Float16)acc[r];
        *(h8*)(Vt + ((size_t)(bb * DK + ncol)) * T_ + t0) = vals;
    } else {
        _Float16* O = (z == 0) ? Kh : Qh;
#pragma unroll
        for (int r = 0; r < 8; ++r) {
            int m = tileM * 16 + r + 8 * h;
            O[(size_t)m * DK + ncol] = (_Float16)acc[r];
        }
    }
}

// ---------------------------------------------------------------------------
// Kernel 2: fused flash attention. 8 waves/block, each wave = 16 query rows.
// Double-buffered staging: K tile via per-thread ASYNC copies (contiguous 4KB),
// V^T tile via one TDM 2-D descriptor issued by wave 0 (64 rows x 64B, stride 8KB).
// ---------------------------------------------------------------------------
__global__ __launch_bounds__(256) void flash_kernel(
    const _Float16* __restrict__ Qh, const _Float16* __restrict__ Kh,
    const _Float16* __restrict__ Vt, const int* __restrict__ pad,
    float* __restrict__ out)
{
    __shared__ _Float16 ldsK[2 * 32 * 64];     // [buf][key][feat]
    __shared__ _Float16 ldsVt[2 * 64 * 32];    // [buf][feat][key]
    __shared__ _Float16 ldsP[8][16 * 32];      // per-wave P staging [row][key]

    const int b     = blockIdx.y;
    const int tid   = threadIdx.x;
    const int lane  = tid & 31;
    const int wave  = tid >> 5;
    const int h     = lane >> 4;
    const int mn    = lane & 15;
    const int qrow0 = blockIdx.x * 128 + wave * 16;
    const float scale = 0.044194173824159216f;   // 1/sqrt(512)

    // K staging: thread tid moves 16B chunk tid of the contiguous 4KB tile
    _Float16* dstK0 = ldsK + tid * 8;
    const _Float16* srcKb = Kh + ((size_t)b * T_) * DK + tid * 8;
    // V^T staging source: per-batch slab [64][4096] halves
    const uint64_t gVb = (uint64_t)(uintptr_t)(Vt + ((size_t)b * DK) * T_);

    // persistent Q A-fragments (K-dim 64 -> two 16x32 fragments)
    const _Float16* qrow = Qh + ((size_t)b * T_ + qrow0 + mn) * DK;
    v16h aq0 = make_frag(((const h8*)(qrow + 0  + h * 8))[0],
                         ((const h8*)(qrow + 16 + h * 8))[0]);
    v16h aq1 = make_frag(((const h8*)(qrow + 32 + h * 8))[0],
                         ((const h8*)(qrow + 48 + h * 8))[0]);

    float mrow[8], lrow[8];
    v8f oacc[4];
    v8f vzero = {};
#pragma unroll
    for (int r = 0; r < 8; ++r) { mrow[r] = -1e30f; lrow[r] = 0.f; }
#pragma unroll
    for (int c = 0; c < 4; ++c) oacc[c] = vzero;

    const int* pmask = pad + (size_t)b * T_;

    // prologue: stage tile 0 into buffer 0
    async_b128(dstK0, srcKb);
    if (wave == 0)
        tdm_load_2d((uint32_t)(uintptr_t)ldsVt, gVb + 0,
                    T_, DK, 32, DK, T_);

    for (int it = 0; it < T_ / 32; ++it) {
        const int buf = it & 1;
        const int s0  = it * 32;

        // own counters drained, then block-wide publish via barrier
        asm volatile("s_wait_asynccnt 0x0" ::: "memory");
        if (wave == 0) __builtin_amdgcn_s_wait_tensorcnt(0);
        __syncthreads();

        // prefetch tile it+1 into the other buffer (overlaps compute below);
        // safe: the barrier above means every wave finished reading buf^1
        if (s0 + 32 < T_) {
            int nb = buf ^ 1;
            async_b128(dstK0 + nb * 2048, srcKb + (size_t)(s0 + 32) * DK);
            if (wave == 0)
                tdm_load_2d((uint32_t)(uintptr_t)(ldsVt + nb * 2048),
                            gVb + (uint64_t)(s0 + 32) * 2,
                            T_, DK, 32, DK, T_);
        }

        const _Float16* lk = ldsK  + buf * 2048;
        const _Float16* lv = ldsVt + buf * 2048;

        // S = Q K^T : two 16x16 D-tiles (j = key sub-tile)
        v8f s[2];
#pragma unroll
        for (int j = 0; j < 2; ++j) {
            const h8* bk0 = (const h8*)&lk[(j * 16 + mn) * 64 + 0  + h * 8];
            const h8* bk1 = (const h8*)&lk[(j * 16 + mn) * 64 + 32 + h * 8];
            v16h b0 = make_frag(bk0[0], bk0[2]);
            v16h b1 = make_frag(bk1[0], bk1[2]);
            v8f sc = vzero;
            sc = __builtin_amdgcn_wmma_f32_16x16x32_f16(false, aq0, false, b0,
                                                        (short)0, sc, false, false);
            sc = __builtin_amdgcn_wmma_f32_16x16x32_f16(false, aq1, false, b1,
                                                        (short)0, sc, false, false);
            s[j] = sc;
        }

        // mask + online softmax (rows live across 16-lane halves -> xor 1,2,4,8)
        const bool msk0 = (pmask[s0 + mn] == 1);
        const bool msk1 = (pmask[s0 + 16 + mn] == 1);
#pragma unroll
        for (int r = 0; r < 8; ++r) {
            float s0v = msk0 ? -__builtin_inff() : s[0][r] * scale;
            float s1v = msk1 ? -__builtin_inff() : s[1][r] * scale;
            float t = fmaxf(s0v, s1v);
#pragma unroll
            for (int d = 1; d < 16; d <<= 1) t = fmaxf(t, __shfl_xor(t, d, 32));
            float mnew  = fmaxf(mrow[r], t);
            float alpha = __expf(mrow[r] - mnew);
            float p0 = __expf(s0v - mnew);
            float p1 = __expf(s1v - mnew);
            float rs = p0 + p1;
#pragma unroll
            for (int d = 1; d < 16; d <<= 1) rs += __shfl_xor(rs, d, 32);
            lrow[r] = lrow[r] * alpha + rs;
            mrow[r] = mnew;
#pragma unroll
            for (int c = 0; c < 4; ++c) oacc[c][r] *= alpha;
            int prow = r + 8 * h;
            ldsP[wave][prow * 32 + mn]      = (_Float16)p0;
            ldsP[wave][prow * 32 + 16 + mn] = (_Float16)p1;
        }

        // per-wave LDS ops are in-order; drain stores before re-reading
        asm volatile("s_wait_dscnt 0" ::: "memory");

        // reload P in A-layout (elems 0..7 -> key h*8.., 8..15 -> key 16+h*8..)
        const h8* pp0 = (const h8*)&ldsP[wave][mn * 32 + h * 8];
        const h8* pp1 = (const h8*)&ldsP[wave][mn * 32 + 16 + h * 8];
        v16h ap = make_frag(pp0[0], pp1[0]);

        // O += P V : B fragments straight out of the transposed V tile
#pragma unroll
        for (int c = 0; c < 4; ++c) {
            const h8* bv = (const h8*)&lv[(c * 16 + mn) * 32 + h * 8];
            v16h bvv = make_frag(bv[0], bv[2]);
            oacc[c] = __builtin_amdgcn_wmma_f32_16x16x32_f16(false, ap, false, bvv,
                                                             (short)0, oacc[c], false, false);
        }
    }

    // epilogue: divide by l, store fp32
    float* obase = out + ((size_t)b * T_ + qrow0) * DK;
#pragma unroll
    for (int c = 0; c < 4; ++c) {
#pragma unroll
        for (int r = 0; r < 8; ++r) {
            int m = r + 8 * h;
            float l = lrow[r];
            float val = (l > 0.f) ? oacc[c][r] / l : 0.f;
            obase[(size_t)m * DK + c * 16 + mn] = val;
        }
    }
}

// ---------------------------------------------------------------------------
extern "C" void kernel_launch(void* const* d_in, const int* in_sizes, int n_in,
                              void* d_out, int out_size, void* d_ws, size_t ws_size,
                              hipStream_t stream) {
    const float* k   = (const float*)d_in[0];
    const float* v   = (const float*)d_in[1];
    const float* q   = (const float*)d_in[2];
    const int*   pad = (const int*)  d_in[3];
    const float* Wk  = (const float*)d_in[4];
    const float* Wq  = (const float*)d_in[5];
    const float* Wv  = (const float*)d_in[6];
    float* out = (float*)d_out;

    const size_t per = (size_t)B_ * T_ * DK;       // halves per projected tensor
    _Float16* Wt = (_Float16*)d_ws;                // 3*64*512 halves
    _Float16* Kh = Wt + (size_t)3 * DK * DM;
    _Float16* Qh = Kh + per;
    _Float16* Vt = Qh + per;                       // transposed V: [b][feat][t]

    wcvt_kernel<<<dim3(3), 256, 0, stream>>>(Wk, Wq, Wv, Wt);

    dim3 pgrid(2048, 1, 3);                        // 8192 wave-tiles / 4 waves
    proj_kernel<<<pgrid, 128, 0, stream>>>(k, q, v, Wt, Kh, Qh, Vt);

    dim3 fgrid(T_ / 128, B_);                      // 8 waves/block, 128 q-rows
    flash_kernel<<<fgrid, 256, 0, stream>>>(Qh, Kh, Vt, pad, out);
}
